// HierarchicalPooler_50242527428620
// MI455X (gfx1250) — compile-verified, matrix operand order checked by reading, not run on hardware
//
#include <hip/hip_runtime.h>
#include <hip/hip_bf16.h>
#include <cstdint>
#include <cstddef>

// ---------------------------------------------------------------------------
// Problem constants (from reference)
// ---------------------------------------------------------------------------
#define N_FINE   10000
#define E_FINE   60000
#define N_COARSE 5000
#define NC_PAD   5008      // 313 * 16
#define E_COARSE 30000
#define KNN      6
#define HID      128
#define NB       5
#define LDS_PAD  8         // bf16 elements of row padding (stride = K+8)

typedef __bf16 bf16_t;
typedef __attribute__((ext_vector_type(16))) __bf16        v16bf;
typedef __attribute__((ext_vector_type(8)))  float         v8f;
typedef __attribute__((ext_vector_type(8)))  unsigned int  v8u;

// ---------------------------------------------------------------------------
// WMMA fragment loads: 16-bit A-matrix 16x32 per-lane layout (CDNA5 ISA)
//   VGPR v<4 : K = 2v, 2v+1  (+8*half);  v>=4 : K = 16+2(v-4) (+8*half)
// Weights stored transposed ([N][Kpad] row-major) so B uses the same path.
// ---------------------------------------------------------------------------
__device__ __forceinline__ v16bf load_frag_p(const bf16_t* __restrict__ rowp,
                                             int kbase, int half) {
  const unsigned int* p = reinterpret_cast<const unsigned int*>(rowp);
  v8u tmp;
#pragma unroll
  for (int v = 0; v < 8; ++v) {
    int kk = ((v < 4) ? 2 * v : 16 + 2 * (v - 4)) + 8 * half + kbase;
    tmp[v] = p[kk >> 1];               // one dword = two bf16 (K, K+1)
  }
  return __builtin_bit_cast(v16bf, tmp);
}

__device__ __forceinline__ v16bf load_frag(const bf16_t* __restrict__ base,
                                           int ld, int row, int kbase, int half) {
  return load_frag_p(base + (size_t)row * ld, kbase, half);
}

// fp32 source row -> bf16 fragment (for aggregated features)
__device__ __forceinline__ v16bf load_frag_f32(const float* __restrict__ rowp,
                                               int kbase, int half) {
  v8u tmp;
#pragma unroll
  for (int v = 0; v < 8; ++v) {
    int kk = ((v < 4) ? 2 * v : 16 + 2 * (v - 4)) + 8 * half + kbase;
    unsigned lo = (unsigned)__builtin_bit_cast(unsigned short, (bf16_t)rowp[kk]);
    unsigned hi = (unsigned)__builtin_bit_cast(unsigned short, (bf16_t)rowp[kk + 1]);
    tmp[v] = lo | (hi << 16);
  }
  return __builtin_bit_cast(v16bf, tmp);
}

// cooperative stage of WT [128 x K] (bf16) into LDS with +LDS_PAD row padding
__device__ __forceinline__ void stage_weights(bf16_t* sW, const bf16_t* __restrict__ WT,
                                              int K) {
  const uint4* g = reinterpret_cast<const uint4*>(WT);
  int cpr = K >> 3;                        // 16B chunks per row
  for (int i = threadIdx.x; i < 128 * cpr; i += 256) {
    int r = i / cpr, c = i - r * cpr;
    *reinterpret_cast<uint4*>(sW + (size_t)r * (K + LDS_PAD) + c * 8) =
        g[(size_t)r * cpr + c];
  }
  __syncthreads();
}

// ---------------------------------------------------------------------------
// Shared full-row (16x128) epilogue.
// C layout: acc[j][r] holds row m = tm*16 + r + 8*half, col = j*16 + (lane&15).
// If g != null: bias -> LayerNorm (width-16 xor-shuffle reductions, halves are
// disjoint row groups) -> optional ln store, residual, f32/bf16 stores.
// ---------------------------------------------------------------------------
__device__ __forceinline__ void epilogue_row(
    v8f (&acc)[8], int tm, int half, int r16,
    const float* __restrict__ bias, int relu,
    const float* __restrict__ g, const float* __restrict__ be,
    const float* __restrict__ base,
    float* __restrict__ out_ln, float* __restrict__ out_f32,
    bf16_t* __restrict__ out_b16) {
  float bv[8], gv[8], bev[8];
#pragma unroll
  for (int j = 0; j < 8; ++j) bv[j] = bias ? bias[j * 16 + r16] : 0.0f;
  if (g) {
#pragma unroll
    for (int j = 0; j < 8; ++j) {
      gv[j]  = g[j * 16 + r16];
      bev[j] = be[j * 16 + r16];
    }
  }
#pragma unroll
  for (int r = 0; r < 8; ++r) {
    int m = tm * 16 + r + 8 * half;
    if (!g) {
#pragma unroll
      for (int j = 0; j < 8; ++j) {
        float v = acc[j][r] + bv[j];
        if (relu) v = fmaxf(v, 0.0f);
        size_t idx = (size_t)m * HID + j * 16 + r16;
        if (out_f32) out_f32[idx] = v;
        if (out_b16) out_b16[idx] = (bf16_t)v;
      }
    } else {
      float vv[8], s = 0.f, ss = 0.f;
#pragma unroll
      for (int j = 0; j < 8; ++j) {
        float v = acc[j][r] + bv[j];
        vv[j] = v; s += v; ss += v * v;
      }
#pragma unroll
      for (int off = 8; off; off >>= 1) {    // reduce across the 16-lane half
        s  += __shfl_xor(s,  off, 16);
        ss += __shfl_xor(ss, off, 16);
      }
      float mu  = s * (1.f / 128.f);
      float var = ss * (1.f / 128.f) - mu * mu;
      float inv = rsqrtf(var + 1e-5f);
#pragma unroll
      for (int j = 0; j < 8; ++j) {
        size_t idx = (size_t)m * HID + j * 16 + r16;
        float ln = (vv[j] - mu) * inv * gv[j] + bev[j];
        if (out_ln) out_ln[idx] = ln;
        float rr = (base ? base[idx] : 0.f) + ln;
        if (out_f32) out_f32[idx] = rr;
        if (out_b16) out_b16[idx] = (bf16_t)rr;
      }
    }
  }
}

// ---------------------------------------------------------------------------
// Generic full-row GEMM: one wave -> 16 rows x 128 cols; 8 waves/block.
// A: bf16 [M, lda]; WT: bf16 [128, K], staged through LDS (K <= 128 here).
// ---------------------------------------------------------------------------
__global__ __launch_bounds__(256)
void k_gemm_row(const bf16_t* __restrict__ A, int lda,
                const bf16_t* __restrict__ WT,
                const float* __restrict__ bias, int relu,
                const float* __restrict__ g, const float* __restrict__ be,
                const float* __restrict__ base,
                float* __restrict__ out_ln, float* __restrict__ out_f32,
                bf16_t* __restrict__ out_b16,
                int Mtiles, int K) {
  __shared__ __align__(16) bf16_t sW[128 * (128 + LDS_PAD)];
  stage_weights(sW, WT, K);

  int wave = threadIdx.x >> 5;
  int lane = threadIdx.x & 31;
  int tm = blockIdx.x * 8 + wave;
  if (tm >= Mtiles) return;
  int half = lane >> 4;
  int r16  = lane & 15;
  const bf16_t* arow = A + (size_t)(tm * 16 + r16) * lda;
  int ldw = K + LDS_PAD;

  v8f acc[8] = {};
  for (int k = 0; k < K; k += 32) {
    v16bf af = load_frag_p(arow, k, half);
#pragma unroll
    for (int j = 0; j < 8; ++j) {
      v16bf bf = load_frag(sW, ldw, j * 16 + r16, k, half);
      acc[j] = __builtin_amdgcn_wmma_f32_16x16x32_bf16(
          false, af, false, bf, (short)0, acc[j], false, false);
    }
  }
  epilogue_row(acc, tm, half, r16, bias, relu, g, be, base, out_ln, out_f32, out_b16);
}

// ---------------------------------------------------------------------------
// Fused edge-MLP layer 1: A row e = [eh[e] | xh[src[e]] | xh[dst[e]]], K=384.
// Gather-concat happens in the fragment loader (segment constant per K-step).
// ---------------------------------------------------------------------------
__global__ __launch_bounds__(256)
void k_gemm_edge1(const bf16_t* __restrict__ eh, const bf16_t* __restrict__ xh,
                  const int* __restrict__ src, const int* __restrict__ dst,
                  const bf16_t* __restrict__ WT, const float* __restrict__ bias,
                  bf16_t* __restrict__ out, int Mtiles) {
  __shared__ __align__(16) bf16_t sW[128 * (384 + LDS_PAD)];
  stage_weights(sW, WT, 384);

  int wave = threadIdx.x >> 5;
  int lane = threadIdx.x & 31;
  int tm = blockIdx.x * 8 + wave;
  if (tm >= Mtiles) return;
  int half = lane >> 4;
  int r16  = lane & 15;
  int e = tm * 16 + r16;
  const bf16_t* rowp[3] = {
      eh + (size_t)e * HID,
      xh + (size_t)src[e] * HID,
      xh + (size_t)dst[e] * HID };
  const int ldw = 384 + LDS_PAD;

  v8f acc[8] = {};
#pragma unroll
  for (int ks = 0; ks < 12; ++ks) {          // 12 * 32 = 384
    int seg = ks >> 2;
    int kl  = (ks & 3) * 32;
    v16bf af = load_frag_p(rowp[seg], kl, half);
#pragma unroll
    for (int j = 0; j < 8; ++j) {
      v16bf bf = load_frag(sW, ldw, j * 16 + r16, ks * 32, half);
      acc[j] = __builtin_amdgcn_wmma_f32_16x16x32_bf16(
          false, af, false, bf, (short)0, acc[j], false, false);
    }
  }
  epilogue_row(acc, tm, half, r16, bias, 1, nullptr, nullptr, nullptr,
               nullptr, nullptr, out);
}

// ---------------------------------------------------------------------------
// Fused node-MLP layer 1: A row n = [xh[n](bf16) | agg[n](f32 -> bf16)], K=256.
// ---------------------------------------------------------------------------
__global__ __launch_bounds__(256)
void k_gemm_node1(const bf16_t* __restrict__ xh, const float* __restrict__ agg,
                  const bf16_t* __restrict__ WT, const float* __restrict__ bias,
                  bf16_t* __restrict__ out, int Mtiles) {
  __shared__ __align__(16) bf16_t sW[128 * (256 + LDS_PAD)];
  stage_weights(sW, WT, 256);

  int wave = threadIdx.x >> 5;
  int lane = threadIdx.x & 31;
  int tm = blockIdx.x * 8 + wave;
  if (tm >= Mtiles) return;
  int half = lane >> 4;
  int r16  = lane & 15;
  int n = tm * 16 + r16;
  const bf16_t* rowx = xh  + (size_t)n * HID;
  const float*  rowa = agg + (size_t)n * HID;
  const int ldw = 256 + LDS_PAD;

  v8f acc[8] = {};
#pragma unroll
  for (int ks = 0; ks < 8; ++ks) {           // 8 * 32 = 256
    int kl = (ks & 3) * 32;
    v16bf af = (ks < 4) ? load_frag_p(rowx, kl, half)
                        : load_frag_f32(rowa, kl, half);
#pragma unroll
    for (int j = 0; j < 8; ++j) {
      v16bf bf = load_frag(sW, ldw, j * 16 + r16, ks * 32, half);
      acc[j] = __builtin_amdgcn_wmma_f32_16x16x32_bf16(
          false, af, false, bf, (short)0, acc[j], false, false);
    }
  }
  epilogue_row(acc, tm, half, r16, bias, 1, nullptr, nullptr, nullptr,
               nullptr, nullptr, out);
}

// ---------------------------------------------------------------------------
// Narrow-output tile GEMM (decoder final layer, N=16 tile, store cols < ncols)
// ---------------------------------------------------------------------------
__global__ __launch_bounds__(256)
void k_gemm_tile(const bf16_t* __restrict__ A, int lda,
                 const bf16_t* __restrict__ WT, int ldb,
                 const float* __restrict__ bias,
                 float* __restrict__ Cf, int Mtiles, int K, int ncols) {
  int wave = threadIdx.x >> 5;
  int lane = threadIdx.x & 31;
  int tm = blockIdx.x * 8 + wave;
  if (tm >= Mtiles) return;
  int half = lane >> 4;
  int r16  = lane & 15;

  v8f acc = {};
  for (int k = 0; k < K; k += 32) {
    v16bf af = load_frag(A,  lda, tm * 16 + r16, k, half);
    v16bf bf = load_frag(WT, ldb, r16,           k, half);
    acc = __builtin_amdgcn_wmma_f32_16x16x32_bf16(
        false, af, false, bf, (short)0, acc, false, false);
  }
  float bv = bias ? bias[r16] : 0.0f;
#pragma unroll
  for (int r = 0; r < 8; ++r) {
    int m = tm * 16 + r + 8 * half;
    float v = acc[r] + bv;
    if (r16 < ncols) Cf[(size_t)m * ncols + r16] = v;
  }
}

// ---------------------------------------------------------------------------
// Support kernels
// ---------------------------------------------------------------------------
__global__ void k_wT_bf16(const float* __restrict__ W, int K, int N,
                          bf16_t* __restrict__ WT, int Kpad, long long total) {
  long long i = blockIdx.x * (long long)blockDim.x + threadIdx.x;
  for (; i < total; i += gridDim.x * (long long)blockDim.x) {
    int n = (int)(i / Kpad), k = (int)(i % Kpad);
    float v = (k < K && n < N) ? W[(size_t)k * N + n] : 0.0f;
    WT[i] = (bf16_t)v;
  }
}

__global__ void k_pack_bf16(const float* __restrict__ src, int Mvalid, int Kin,
                            int Kpad, bf16_t* __restrict__ dst, long long total) {
  long long i = blockIdx.x * (long long)blockDim.x + threadIdx.x;
  for (; i < total; i += gridDim.x * (long long)blockDim.x) {
    long long m = i / Kpad; int k = (int)(i % Kpad);
    float v = (m < Mvalid && k < Kin) ? src[(size_t)m * Kin + k] : 0.0f;
    dst[i] = (bf16_t)v;
  }
}

__global__ void k_zero(float* __restrict__ p, long long n) {
  long long i = blockIdx.x * (long long)blockDim.x + threadIdx.x;
  for (; i < n; i += gridDim.x * (long long)blockDim.x) p[i] = 0.0f;
}

__global__ void k_scatter_add(const float* __restrict__ ln_e,
                              const int* __restrict__ dst,
                              float* __restrict__ agg, long long E) {
  long long i = blockIdx.x * (long long)blockDim.x + threadIdx.x;
  long long total = E * 128;
  for (; i < total; i += gridDim.x * (long long)blockDim.x) {
    long long e = i / 128; int c = (int)(i % 128);
    atomicAdd(&agg[(size_t)dst[e] * 128 + c], ln_e[i]);
  }
}

__global__ void k_gather_rows(const float* __restrict__ x,
                              const int* __restrict__ perm,
                              bf16_t* __restrict__ out, int Mvalid, long long total) {
  long long i = blockIdx.x * (long long)blockDim.x + threadIdx.x;
  for (; i < total; i += gridDim.x * (long long)blockDim.x) {
    long long n = i / 128; int c = (int)(i % 128);
    float v = (n < Mvalid) ? x[(size_t)perm[n] * 128 + c] : 0.0f;
    out[i] = (bf16_t)v;
  }
}

__global__ void k_gather_pos(const float* __restrict__ pos,
                             const int* __restrict__ perm,
                             float* __restrict__ pos_c, int NC) {
  int i = blockIdx.x * blockDim.x + threadIdx.x;
  if (i < NC * 3) {
    int n = i / 3, d = i % 3;
    pos_c[i] = pos[(size_t)perm[n] * 3 + d];
  }
}

__global__ void k_coarse_eattr(const float* __restrict__ pos,
                               const int* __restrict__ perm,
                               const int* __restrict__ srcc,
                               const int* __restrict__ dstc,
                               bf16_t* __restrict__ out, int EC) {
  int e = blockIdx.x * blockDim.x + threadIdx.x;
  if (e >= EC) return;
  int a = perm[srcc[e]], b = perm[dstc[e]];
  float rx = pos[(size_t)b * 3 + 0] - pos[(size_t)a * 3 + 0];
  float ry = pos[(size_t)b * 3 + 1] - pos[(size_t)a * 3 + 1];
  float rz = pos[(size_t)b * 3 + 2] - pos[(size_t)a * 3 + 2];
  float nr = sqrtf(rx * rx + ry * ry + rz * rz);
  bf16_t* o = out + (size_t)e * 32;
  o[0] = (bf16_t)rx; o[1] = (bf16_t)ry; o[2] = (bf16_t)rz; o[3] = (bf16_t)nr;
#pragma unroll
  for (int k = 4; k < 32; ++k) o[k] = (bf16_t)0.0f;
}

__global__ __launch_bounds__(256)
void k_upsample(const float* __restrict__ pos, const float* __restrict__ pos_c,
                const int* __restrict__ knn, const float* __restrict__ x_c,
                bf16_t* __restrict__ up_b, int N) {
  int n = blockIdx.x * 8 + (threadIdx.x >> 5);
  if (n >= N) return;
  int lane = threadIdx.x & 31;
  float px = pos[(size_t)n * 3], py = pos[(size_t)n * 3 + 1], pz = pos[(size_t)n * 3 + 2];
  float w[KNN], wsum = 0.f;
  int id[KNN];
#pragma unroll
  for (int k = 0; k < KNN; ++k) {
    int c = knn[(size_t)n * KNN + k];
    id[k] = c;
    float dx = px - pos_c[(size_t)c * 3];
    float dy = py - pos_c[(size_t)c * 3 + 1];
    float dz = pz - pos_c[(size_t)c * 3 + 2];
    float wk = 1.0f / (dx * dx + dy * dy + dz * dz + 1e-8f);
    w[k] = wk; wsum += wk;
  }
  float inv = 1.0f / wsum;
#pragma unroll
  for (int i = 0; i < 4; ++i) {
    int c = lane + 32 * i;
    float acc = 0.f;
#pragma unroll
    for (int k = 0; k < KNN; ++k) acc += w[k] * x_c[(size_t)id[k] * 128 + c];
    up_b[(size_t)n * 128 + c] = (bf16_t)(acc * inv);
  }
}

__global__ void k_add_res(const float* __restrict__ a, const float* __restrict__ b,
                          float* __restrict__ of, bf16_t* __restrict__ ob, long long n) {
  long long i = blockIdx.x * (long long)blockDim.x + threadIdx.x;
  for (; i < n; i += gridDim.x * (long long)blockDim.x) {
    float v = a[i] + b[i];
    of[i] = v; ob[i] = (bf16_t)v;
  }
}

__global__ void k_pad_bias(const float* __restrict__ b, float* __restrict__ out,
                           int n, int npad) {
  int i = threadIdx.x;
  if (i < npad) out[i] = (i < n) ? b[i] : 0.0f;
}

// ===========================================================================
// Host-side orchestration
// ===========================================================================
namespace {

struct WS {
  char* p; size_t off;
  void* alloc(size_t bytes) {
    void* r = p + off;
    off += (bytes + 255) & ~(size_t)255;
    return r;
  }
};

struct MLPW {
  bf16_t *W1T, *W2T, *W3T;
  const float *b1, *b2, *b3, *g, *be;
  int K1;
};

inline int gsz(long long total) {
  long long b = (total + 255) / 256;
  return (int)(b > 65535 ? 65535 : b);
}

bf16_t* convW(WS& ws, const float* W, int K, int N, int Kpad, int Nrows, hipStream_t s) {
  bf16_t* WT = (bf16_t*)ws.alloc((size_t)Nrows * Kpad * sizeof(bf16_t));
  long long total = (long long)Nrows * Kpad;
  k_wT_bf16<<<gsz(total), 256, 0, s>>>(W, K, N, WT, Kpad, total);
  return WT;
}

// params leaves are jax-pytree sorted: W1,W2,W3,b1,b2,b3[,be,g]
MLPW make_mlpw(WS& ws, void* const* din, int base, int K1, int K1pad,
               int Nout3, int N3rows, bool ln, hipStream_t s) {
  MLPW m;
  m.W1T = convW(ws, (const float*)din[base + 0], K1,  HID, K1pad, HID,    s);
  m.W2T = convW(ws, (const float*)din[base + 1], HID, HID, HID,   HID,    s);
  m.W3T = convW(ws, (const float*)din[base + 2], HID, Nout3, HID, N3rows, s);
  m.b1 = (const float*)din[base + 3];
  m.b2 = (const float*)din[base + 4];
  m.b3 = (const float*)din[base + 5];
  m.be = ln ? (const float*)din[base + 6] : nullptr;
  m.g  = ln ? (const float*)din[base + 7] : nullptr;
  m.K1 = K1pad;
  return m;
}

} // namespace

extern "C" void kernel_launch(void* const* d_in, const int* in_sizes, int n_in,
                              void* d_out, int out_size, void* d_ws, size_t ws_size,
                              hipStream_t stream) {
  (void)in_sizes; (void)n_in; (void)out_size; (void)ws_size;

  // ------ data inputs (setup_inputs dict order) ------
  const float* x_in  = (const float*)d_in[0];   // [N,12]
  const float* pos   = (const float*)d_in[1];   // [N,3]
  const float* eattr = (const float*)d_in[2];   // [E,4]
  const int*   eidx  = (const int*)  d_in[3];   // [2,E]
  const int*   perm  = (const int*)  d_in[4];   // [NC]
  const int*   ceidx = (const int*)  d_in[5];   // [2,EC]
  const int*   knn   = (const int*)  d_in[6];   // [N,K]
  const int* src  = eidx,  * dst  = eidx  + E_FINE;
  const int* csrc = ceidx, * cdst = ceidx + E_COARSE;

  // ------ params leaves (jax pytree sorted keys) ------
  const int COARSE = 7;
  const int DEC    = COARSE + 80;
  const int DOWN   = DEC + 6;
  const int DLW    = DOWN + 80;
  const int DLB    = DLW + 1;
  const int EENC   = DLB + 1;
  const int NENC   = EENC + 8;
  const int UP     = NENC + 8;
  const int ULW    = UP + 80;
  const int ULB    = ULW + 1;

  WS ws{(char*)d_ws, 0};
  hipStream_t s = stream;

  // ------ weight conversion (bf16, transposed, K-padded) ------
  MLPW node_enc = make_mlpw(ws, d_in, NENC, 12, 32, HID, HID, true, s);
  MLPW edge_enc = make_mlpw(ws, d_in, EENC, 4,  32, HID, HID, true, s);
  MLPW down_e[NB], down_n[NB], coarse_e[NB], coarse_n[NB], up_e[NB], up_n[NB];
  for (int i = 0; i < NB; ++i) {
    down_e[i]   = make_mlpw(ws, d_in, DOWN   + 16 * i,     3 * HID, 3 * HID, HID, HID, true, s);
    down_n[i]   = make_mlpw(ws, d_in, DOWN   + 16 * i + 8, 2 * HID, 2 * HID, HID, HID, true, s);
    coarse_e[i] = make_mlpw(ws, d_in, COARSE + 16 * i,     3 * HID, 3 * HID, HID, HID, true, s);
    coarse_n[i] = make_mlpw(ws, d_in, COARSE + 16 * i + 8, 2 * HID, 2 * HID, HID, HID, true, s);
    up_e[i]     = make_mlpw(ws, d_in, UP     + 16 * i,     3 * HID, 3 * HID, HID, HID, true, s);
    up_n[i]     = make_mlpw(ws, d_in, UP     + 16 * i + 8, 2 * HID, 2 * HID, HID, HID, true, s);
  }
  MLPW dec = make_mlpw(ws, d_in, DEC, HID, HID, 3, 16, false, s);
  bf16_t* dlinT = convW(ws, (const float*)d_in[DLW], HID, HID, HID, HID, s);
  bf16_t* ulinT = convW(ws, (const float*)d_in[ULW], HID, HID, HID, HID, s);
  const float* dlb = (const float*)d_in[DLB];
  const float* ulb = (const float*)d_in[ULB];
  float* b3pad = (float*)ws.alloc(16 * sizeof(float));

  // ------ activation buffers ------
  auto af = [&](long long n) { return (float*) ws.alloc((size_t)n * sizeof(float)); };
  auto ab = [&](long long n) { return (bf16_t*)ws.alloc((size_t)n * sizeof(bf16_t)); };
  bf16_t* Xin_b  = ab((long long)N_FINE * 32);
  bf16_t* Ein_b  = ab((long long)E_FINE * 32);
  float*  xh     = af((long long)N_FINE * HID);   bf16_t* xh_b  = ab((long long)N_FINE * HID);
  float*  eh     = af((long long)E_FINE * HID);   bf16_t* eh_b  = ab((long long)E_FINE * HID);
  bf16_t* h1     = ab((long long)E_FINE * HID);
  bf16_t* h2     = ab((long long)E_FINE * HID);
  float*  ln_e   = af((long long)E_FINE * HID);
  float*  agg    = af((long long)N_FINE * HID);
  bf16_t* pool_b = ab((long long)NC_PAD * HID);
  float*  x_c    = af((long long)NC_PAD * HID);   bf16_t* x_c_b = ab((long long)NC_PAD * HID);
  float*  e_c    = af((long long)E_COARSE * HID); bf16_t* e_c_b = ab((long long)E_COARSE * HID);
  bf16_t* eatt_b = ab((long long)E_COARSE * 32);
  float*  pos_c  = af((long long)N_COARSE * 3);
  bf16_t* up_b   = ab((long long)N_FINE * HID);
  float*  gout   = af((long long)N_FINE * HID);   // upsample-linear output
  float*  x_f    = af((long long)N_FINE * HID);   bf16_t* x_f_b = ab((long long)N_FINE * HID);
  float*  out_f  = (float*)d_out;                 // [N,3]

  const int MT_N  = N_FINE / 16;    // 625
  const int MT_E  = E_FINE / 16;    // 3750
  const int MT_NC = NC_PAD / 16;    // 313
  const int MT_EC = E_COARSE / 16;  // 1875

  auto row = [&](const bf16_t* A, int lda, const bf16_t* WT, const float* bias,
                 int relu, const float* g, const float* be, const float* base,
                 float* oln, float* of, bf16_t* ob, int Mtiles, int K) {
    k_gemm_row<<<dim3((Mtiles + 7) / 8), 256, 0, s>>>(
        A, lda, WT, bias, relu, g, be, base, oln, of, ob, Mtiles, K);
  };
  // layers 2+3 of an MLP: h -> h(relu) -> fused-LN output
  auto tail = [&](const MLPW& m, int Mtiles, const float* base,
                  float* oln, float* of, bf16_t* ob) {
    row(h1, HID, m.W2T, m.b2, 1, nullptr, nullptr, nullptr, nullptr, nullptr, h2,
        Mtiles, HID);
    row(h2, HID, m.W3T, m.b3, 0, m.g, m.be, base, oln, of, ob, Mtiles, HID);
  };

  // ------ encoders ------
  {
    long long t = (long long)N_FINE * 32;
    k_pack_bf16<<<gsz(t), 256, 0, s>>>(x_in, N_FINE, 12, 32, Xin_b, t);
    row(Xin_b, 32, node_enc.W1T, node_enc.b1, 1, nullptr, nullptr, nullptr,
        nullptr, nullptr, h1, MT_N, 32);
    tail(node_enc, MT_N, nullptr, nullptr, xh, xh_b);

    t = (long long)E_FINE * 32;
    k_pack_bf16<<<gsz(t), 256, 0, s>>>(eattr, E_FINE, 4, 32, Ein_b, t);
    row(Ein_b, 32, edge_enc.W1T, edge_enc.b1, 1, nullptr, nullptr, nullptr,
        nullptr, nullptr, h1, MT_E, 32);
    tail(edge_enc, MT_E, nullptr, nullptr, eh, eh_b);
  }

  // ------ fine message passing (down blocks) ------
  for (int i = 0; i < NB; ++i) {
    k_gemm_edge1<<<dim3((MT_E + 7) / 8), 256, 0, s>>>(
        eh_b, xh_b, src, dst, down_e[i].W1T, down_e[i].b1, h1, MT_E);
    tail(down_e[i], MT_E, eh, ln_e, eh, eh_b);          // ln for agg + residual
    k_zero<<<gsz((long long)N_FINE * HID), 256, 0, s>>>(agg, (long long)N_FINE * HID);
    k_scatter_add<<<gsz((long long)E_FINE * HID), 256, 0, s>>>(ln_e, dst, agg, E_FINE);
    k_gemm_node1<<<dim3((MT_N + 7) / 8), 256, 0, s>>>(
        xh_b, agg, down_n[i].W1T, down_n[i].b1, h1, MT_N);
    tail(down_n[i], MT_N, xh, nullptr, xh, xh_b);       // node residual
  }
  // x_skip == xh, e_skip == eh/eh_b (frozen from here on)

  // ------ downsample ------
  k_gather_rows<<<gsz((long long)NC_PAD * HID), 256, 0, s>>>(
      xh, perm, pool_b, N_COARSE, (long long)NC_PAD * HID);
  row(pool_b, HID, dlinT, dlb, 0, nullptr, nullptr, nullptr,
      nullptr, x_c, x_c_b, MT_NC, HID);
  k_gather_pos<<<(N_COARSE * 3 + 255) / 256, 256, 0, s>>>(pos, perm, pos_c, N_COARSE);
  k_coarse_eattr<<<(E_COARSE + 255) / 256, 256, 0, s>>>(pos, perm, csrc, cdst, eatt_b, E_COARSE);
  row(eatt_b, 32, edge_enc.W1T, edge_enc.b1, 1, nullptr, nullptr, nullptr,
      nullptr, nullptr, h1, MT_EC, 32);
  tail(edge_enc, MT_EC, nullptr, nullptr, e_c, e_c_b);

  // ------ coarse message passing ------
  for (int i = 0; i < NB; ++i) {
    k_gemm_edge1<<<dim3((MT_EC + 7) / 8), 256, 0, s>>>(
        e_c_b, x_c_b, csrc, cdst, coarse_e[i].W1T, coarse_e[i].b1, h1, MT_EC);
    tail(coarse_e[i], MT_EC, e_c, ln_e, e_c, e_c_b);
    k_zero<<<gsz((long long)NC_PAD * HID), 256, 0, s>>>(agg, (long long)NC_PAD * HID);
    k_scatter_add<<<gsz((long long)E_COARSE * HID), 256, 0, s>>>(ln_e, cdst, agg, E_COARSE);
    k_gemm_node1<<<dim3((MT_NC + 7) / 8), 256, 0, s>>>(
        x_c_b, agg, coarse_n[i].W1T, coarse_n[i].b1, h1, MT_NC);
    tail(coarse_n[i], MT_NC, x_c, nullptr, x_c, x_c_b);
  }

  // ------ upsample: inverse-distance kNN interp + linear + skip ------
  k_upsample<<<dim3((N_FINE + 7) / 8), 256, 0, s>>>(pos, pos_c, knn, x_c, up_b, N_FINE);
  row(up_b, HID, ulinT, ulb, 0, nullptr, nullptr, nullptr,
      nullptr, gout, nullptr, MT_N, HID);
  k_add_res<<<gsz((long long)N_FINE * HID), 256, 0, s>>>(
      gout, xh, x_f, x_f_b, (long long)N_FINE * HID);

  // ------ fine message passing (up blocks; edge input = frozen e_skip) ------
  for (int i = 0; i < NB; ++i) {
    k_gemm_edge1<<<dim3((MT_E + 7) / 8), 256, 0, s>>>(
        eh_b, x_f_b, src, dst, up_e[i].W1T, up_e[i].b1, h1, MT_E);
    tail(up_e[i], MT_E, nullptr, ln_e, nullptr, nullptr);  // edge result discarded
    k_zero<<<gsz((long long)N_FINE * HID), 256, 0, s>>>(agg, (long long)N_FINE * HID);
    k_scatter_add<<<gsz((long long)E_FINE * HID), 256, 0, s>>>(ln_e, dst, agg, E_FINE);
    k_gemm_node1<<<dim3((MT_N + 7) / 8), 256, 0, s>>>(
        x_f_b, agg, up_n[i].W1T, up_n[i].b1, h1, MT_N);
    tail(up_n[i], MT_N, x_f, nullptr, x_f, x_f_b);
  }

  // ------ decoder (no LN): 128 -> 128 -> 128 -> 3 ------
  row(x_f_b, HID, dec.W1T, dec.b1, 1, nullptr, nullptr, nullptr,
      nullptr, nullptr, h1, MT_N, HID);
  row(h1, HID, dec.W2T, dec.b2, 1, nullptr, nullptr, nullptr,
      nullptr, nullptr, h2, MT_N, HID);
  k_pad_bias<<<1, 16, 0, s>>>(dec.b3, b3pad, 3, 16);
  k_gemm_tile<<<dim3((MT_N + 7) / 8), 256, 0, s>>>(
      h2, HID, dec.W3T, HID, b3pad, out_f, MT_N, HID, 3);
}
